// InactiveEdgeLayer_85744727097812
// MI455X (gfx1250) — matrix operand order repaired and checked by reading, not compiled, not run on hardware
//
#include <hip/hip_runtime.h>
#include <cmath>

typedef _Float16 h16;
typedef _Float16 v16h __attribute__((ext_vector_type(16)));
typedef _Float16 v8h  __attribute__((ext_vector_type(8)));
typedef float    v8f  __attribute__((ext_vector_type(8)));

#define LN_EPS 1e-8f

//==================== weight pre-packing =====================
// Packs an f32 (K x N) matrix (optionally a vertical concat of two pre-scaled
// sources) into the per-fragment f16 layout consumed by the B operand of
// v_wmma_f32_16x16x32_f16:
//   block = kc * (N/16) + nt  (512 halfs / block)
//   within block: lane(0..31)*16 + i(0..15)
//   K = 32*kc + 8*(lane>>4) + (i&7) + 16*(i>>3) ,  N = 16*nt + (lane&15)
__global__ void pack_weights_kernel(const float* __restrict__ srcA, int K_A,
                                    const float* __restrict__ srcB, int K_B,
                                    float scaleA, float scaleB,
                                    int Ncols, int K_pad,
                                    h16* __restrict__ dst)
{
    int ntiles = Ncols >> 4;
    int total  = (K_pad >> 5) * ntiles * 512;
    for (int t = blockIdx.x * blockDim.x + threadIdx.x; t < total;
         t += gridDim.x * blockDim.x) {
        int i    = t & 15;
        int lane = (t >> 4) & 31;
        int blk  = t >> 9;
        int kc = blk / ntiles;
        int nt = blk - kc * ntiles;
        int K  = 32 * kc + 8 * (lane >> 4) + (i & 7) + 16 * (i >> 3);
        int Nn = 16 * nt + (lane & 15);
        float v = 0.0f;
        if (K < K_A)            v = srcA[K * Ncols + Nn] * scaleA;
        else if (K < K_A + K_B) v = srcB[(K - K_A) * Ncols + Nn] * scaleB;
        dst[t] = (h16)v;
    }
}

//==================== WMMA fragment helpers =====================
// A (16x32 f16): lane holds row M=lane&15; K = 32*kc + 8*(lane>>4) + {0..7, 16..23}
__device__ __forceinline__ v16h load_a_frag(const h16* base, int rowStride, int kBase, int lane)
{
    int m  = lane & 15;
    int hh = lane >> 4;
    const h16* p = base + m * rowStride + kBase + 8 * hh;
    v8h lo = *(const v8h*)(p);
    v8h hi = *(const v8h*)(p + 16);
    v16h r;
#pragma unroll
    for (int i = 0; i < 8; i++) { r[i] = lo[i]; r[i + 8] = hi[i]; }
    return r;
}

// B: pre-packed, 32 contiguous bytes per lane per fragment
__device__ __forceinline__ v16h load_b_frag(const h16* wpk, int blk, int lane)
{
    const h16* p = wpk + ((size_t)blk << 9) + (lane << 4);
    v8h lo = *(const v8h*)(p);
    v8h hi = *(const v8h*)(p + 8);
    v16h r;
#pragma unroll
    for (int i = 0; i < 8; i++) { r[i] = lo[i]; r[i + 8] = hi[i]; }
    return r;
}

__device__ __forceinline__ v8f wmma16(v16h a, v16h b, v8f c)
{
    return __builtin_amdgcn_wmma_f32_16x16x32_f16(false, a, false, b, (short)0, c, false, false);
}

__device__ __forceinline__ v8f vzero8()
{
    v8f z;
#pragma unroll
    for (int i = 0; i < 8; i++) z[i] = 0.0f;
    return z;
}

// D/C layout: VGPR v -> row M = v + 8*(lane>>4), col N = lane&15
__device__ __forceinline__ void store_d(float* base, int rowStride, int colBase, v8f acc, int lane)
{
    int n  = lane & 15;
    int hh = lane >> 4;
#pragma unroll
    for (int v = 0; v < 8; v++)
        base[(v + 8 * hh) * rowStride + colBase + n] = acc[v];
}

__device__ __forceinline__ float sigmoidf_(float x) { return 1.0f / (1.0f + __expf(-x)); }

//==================== main edge kernel =====================
// One wave32 per tile of 16 edges.
// LDS (28672 B), phases overlap dead regions:
//  A-phase: A0 f16[16][480]@0 | A1 f16[16][288]@15360 | A2 f16[16][96]@24576
//  O-phase: O0 f32[16][160]@0 | OP1 f32[16][48]@10240 | ON1@13312 | OP2 f32[16][16]@16384 | ON2@17408
//  C-phase: ScA f16[16][64]@18432 | G1A f16[3][16][32]@20480 | G2A f16[5][16][32]@23552
__global__ __launch_bounds__(32) void edge_layer_kernel(
    const float* __restrict__ latents,
    const float* __restrict__ nodef,
    const float* __restrict__ edgef,
    const float* __restrict__ wD,
    const float* __restrict__ ln_n_w0, const float* __restrict__ ln_n_b0,
    const float* __restrict__ ln_n_w1, const float* __restrict__ ln_n_w2,
    const float* __restrict__ ln_e_w0, const float* __restrict__ ln_e_b0,
    const float* __restrict__ ln_e_w1, const float* __restrict__ ln_e_w2,
    const float* __restrict__ lp_b0v,
    const int*   __restrict__ edge_index,
    const int*   __restrict__ active,
    int E_total, int nA,
    const h16* __restrict__ W0pk,  const h16* __restrict__ W1Ppk,
    const h16* __restrict__ W1Npk, const h16* __restrict__ W2Ppk,
    const h16* __restrict__ W2Npk, const h16* __restrict__ LP0pk,
    const h16* __restrict__ LP1pk, const h16* __restrict__ LP2pk,
    float* __restrict__ out)
{
    __shared__ __align__(16) unsigned char smem[28672];
    h16*   A0  = (h16*)(smem);
    h16*   A1  = (h16*)(smem + 15360);
    h16*   A2  = (h16*)(smem + 24576);
    float* O0  = (float*)(smem);
    float* OP1 = (float*)(smem + 10240);
    float* ON1 = (float*)(smem + 13312);
    float* OP2 = (float*)(smem + 16384);
    float* ON2 = (float*)(smem + 17408);
    h16*   ScA = (h16*)(smem + 18432);
    h16*   G1A = (h16*)(smem + 20480);
    h16*   G2A = (h16*)(smem + 23552);

    const int lane = threadIdx.x;
    const int m    = lane & 15;    // edge within tile (2 lanes per edge)
    const int hh   = lane >> 4;    // half-lane id

    int r  = blockIdx.x * 16 + m;
    int rr = (r < nA) ? r : (nA - 1);
    int aid = active[rr];
    int ec  = edge_index[aid];
    int en  = edge_index[E_total + aid];

    const float* Dg = wD + (size_t)rr * 81;
    float D00 = Dg[0];
    float D1[9];
    float D2[25];
#pragma unroll
    for (int i = 0; i < 3; i++)
#pragma unroll
        for (int j = 0; j < 3; j++) D1[i*3+j] = Dg[(1+i)*9 + (1+j)];
#pragma unroll
    for (int i = 0; i < 5; i++)
#pragma unroll
        for (int j = 0; j < 5; j++) D2[i*5+j] = Dg[(4+i)*9 + (4+j)];

    //---------------- phase A: separable LN + forward Wigner rotation ----------------
    for (int s = 0; s < 3; s++) {
        const float *src, *w0, *b0, *w1, *w2;
        if (s == 1) { src = edgef + (size_t)rr * 240; w0 = ln_e_w0; b0 = ln_e_b0; w1 = ln_e_w1; w2 = ln_e_w2; }
        else        { int nd = (s == 0) ? ec : en;
                      src = nodef + (size_t)nd * 240; w0 = ln_n_w0; b0 = ln_n_b0; w1 = ln_n_w1; w2 = ln_n_w2; }

        // l=0 layernorm over 64 scalars (32 per half-lane)
        float x0l[32];
        float sum = 0.0f;
#pragma unroll
        for (int k = 0; k < 32; k++) { x0l[k] = src[32*hh + k]; sum += x0l[k]; }
        sum += __shfl_xor(sum, 16);
        float mean = sum * (1.0f / 64.0f);
        float ss = 0.0f;
#pragma unroll
        for (int k = 0; k < 32; k++) { x0l[k] -= mean; ss += x0l[k] * x0l[k]; }
        ss += __shfl_xor(ss, 16);
        float inv0 = rsqrtf(ss * (1.0f / 64.0f) + LN_EPS);
#pragma unroll
        for (int k = 0; k < 32; k++) {
            int c = 32*hh + k;
            A0[m*480 + 112*s + c] = (h16)(x0l[k] * inv0 * w0[c] + b0[c]);
        }

        // shared RMS over l=1 (32x3) and l=2 (16x5)
        const float* x1 = src + 64;
        const float* x2 = src + 160;
        float s1 = 0.0f, s2 = 0.0f;
#pragma unroll
        for (int t = 0; t < 48; t++) { float v = x1[48*hh + t]; s1 += v*v; }
#pragma unroll
        for (int t = 0; t < 40; t++) { float v = x2[40*hh + t]; s2 += v*v; }
        float var = s1 * (0.5f / 96.0f) + s2 * (0.5f / 80.0f);
        var += __shfl_xor(var, 16);
        float inv = rsqrtf(var + LN_EPS);

        // l=1: 16 channels per half-lane, rotate with D[1:4,1:4]
#pragma unroll
        for (int cc = 0; cc < 16; cc++) {
            int c = 16*hh + cc;
            float g  = inv * w1[c];
            float v0 = x1[c*3 + 0] * g;
            float v1 = x1[c*3 + 1] * g;
            float v2 = x1[c*3 + 2] * g;
            float r0 = D1[0]*v0 + D1[1]*v1 + D1[2]*v2;
            float r1 = D1[3]*v0 + D1[4]*v1 + D1[5]*v2;
            float r2 = D1[6]*v0 + D1[7]*v1 + D1[8]*v2;
            A0[m*480 + 112*s + 64 + c]  = (h16)r1;   // m0
            A1[m*288 + 48*s + c]        = (h16)r2;   // p1
            A1[m*288 + 144 + 48*s + c]  = (h16)r0;   // n1
        }

        // l=2: 8 channels per half-lane, rotate with D[4:9,4:9]
#pragma unroll
        for (int cc = 0; cc < 8; cc++) {
            int c = 8*hh + cc;
            float g = inv * w2[c];
            float v0 = x2[c*5+0]*g, v1 = x2[c*5+1]*g, v2 = x2[c*5+2]*g;
            float v3 = x2[c*5+3]*g, v4 = x2[c*5+4]*g;
            float rv[5];
#pragma unroll
            for (int i = 0; i < 5; i++)
                rv[i] = D2[i*5+0]*v0 + D2[i*5+1]*v1 + D2[i*5+2]*v2 + D2[i*5+3]*v3 + D2[i*5+4]*v4;
            A0[m*480 + 112*s + 96 + c]      = (h16)rv[2];  // m0
            A1[m*288 + 48*s + 32 + c]       = (h16)rv[3];  // p1
            A1[m*288 + 144 + 48*s + 32 + c] = (h16)rv[1];  // n1
            A2[m*96 + 16*s + c]             = (h16)rv[4];  // p2
            A2[m*96 + 48 + 16*s + c]        = (h16)rv[0];  // n2
        }
    }
    {   // latents + K padding of A0 (cols 464..479)
        const float* lat = latents + (size_t)aid * 128;
#pragma unroll
        for (int k = 0; k < 64; k++) A0[m*480 + 336 + 64*hh + k] = (h16)lat[64*hh + k];
#pragma unroll
        for (int k = 0; k < 8; k++)  A0[m*480 + 464 + 8*hh + k] = (h16)0.0f;
    }
    __syncthreads();

    //---------------- phase B: SO(2) GEMMs via WMMA ----------------
    {   // m0[16x480] @ W0 -> o0[16x160]
        v8f acc[10];
#pragma unroll
        for (int t = 0; t < 10; t++) acc[t] = vzero8();
        for (int kc = 0; kc < 15; kc++) {
            v16h a = load_a_frag(A0, 480, 32*kc, lane);
#pragma unroll
            for (int nt = 0; nt < 10; nt++)
                acc[nt] = wmma16(a, load_b_frag(W0pk, kc*10 + nt, lane), acc[nt]);
        }
        __syncthreads();
#pragma unroll
        for (int nt = 0; nt < 10; nt++) store_d(O0, 160, 16*nt, acc[nt], lane);
    }
    {   // [p1|n1][16x288] @ {[w1r;-w1i],[w1i;w1r]} -> op1,on1[16x48]
        v8f accP[3], accN[3];
#pragma unroll
        for (int t = 0; t < 3; t++) { accP[t] = vzero8(); accN[t] = vzero8(); }
        for (int kc = 0; kc < 9; kc++) {
            v16h a = load_a_frag(A1, 288, 32*kc, lane);
#pragma unroll
            for (int nt = 0; nt < 3; nt++) {
                accP[nt] = wmma16(a, load_b_frag(W1Ppk, kc*3 + nt, lane), accP[nt]);
                accN[nt] = wmma16(a, load_b_frag(W1Npk, kc*3 + nt, lane), accN[nt]);
            }
        }
        __syncthreads();
#pragma unroll
        for (int nt = 0; nt < 3; nt++) {
            store_d(OP1, 48, 16*nt, accP[nt], lane);
            store_d(ON1, 48, 16*nt, accN[nt], lane);
        }
    }
    {   // [p2|n2][16x96] -> op2,on2[16x16]
        v8f accP = vzero8(), accN = vzero8();
        for (int kc = 0; kc < 3; kc++) {
            v16h a = load_a_frag(A2, 96, 32*kc, lane);
            accP = wmma16(a, load_b_frag(W2Ppk, kc, lane), accP);
            accN = wmma16(a, load_b_frag(W2Npk, kc, lane), accN);
        }
        __syncthreads();
        store_d(OP2, 16, 0, accP, lane);
        store_d(ON2, 16, 0, accN, lane);
    }
    __syncthreads();

    //---------------- phase C: inverse rotation, gating, tail staging ----------------
    {
        const float* O0r = O0 + m*160;
        // sc = silu(s0_rot[:64])
#pragma unroll
        for (int k = 0; k < 32; k++) {
            int c = 32*hh + k;
            float x = O0r[c] * D00;
            ScA[m*64 + c] = (h16)(x * sigmoidf_(x));
        }
        // l=1: inverse rotate (D^T) + gate
#pragma unroll
        for (int cc = 0; cc < 16; cc++) {
            int c = 16*hh + cc;
            float g  = sigmoidf_(O0r[64 + c] * D00);
            float v0 = ON1[m*48 + c];
            float v1 = O0r[112 + c];
            float v2 = OP1[m*48 + c];
            float r0 = D1[0]*v0 + D1[3]*v1 + D1[6]*v2;
            float r1 = D1[1]*v0 + D1[4]*v1 + D1[7]*v2;
            float r2 = D1[2]*v0 + D1[5]*v1 + D1[8]*v2;
            G1A[0*512 + m*32 + c] = (h16)(r0 * g);
            G1A[1*512 + m*32 + c] = (h16)(r1 * g);
            G1A[2*512 + m*32 + c] = (h16)(r2 * g);
        }
        // l=2: inverse rotate (D^T) + gate
#pragma unroll
        for (int cc = 0; cc < 8; cc++) {
            int c = 8*hh + cc;
            float g = sigmoidf_(O0r[96 + c] * D00);
            float v0 = ON2[m*16 + c];
            float v1 = ON1[m*48 + 32 + c];
            float v2 = O0r[144 + c];
            float v3 = OP1[m*48 + 32 + c];
            float v4 = OP2[m*16 + c];
#pragma unroll
            for (int i = 0; i < 5; i++) {
                float rv = D2[0*5+i]*v0 + D2[1*5+i]*v1 + D2[2*5+i]*v2 + D2[3*5+i]*v3 + D2[4*5+i]*v4;
                G2A[i*512 + m*32 + c] = (h16)(rv * g);
            }
        }
        // zero K padding (cols 16..31) of G2A
#pragma unroll
        for (int i = 0; i < 5; i++)
#pragma unroll
            for (int t = 0; t < 8; t++)
                G2A[i*512 + m*32 + 16 + 8*hh + t] = (h16)0.0f;
    }
    __syncthreads();

    //---------------- phase D: tail linear layers via WMMA + store ----------------
    const int n = lane & 15;
    {   // y0 = silu(s0) @ lp_w0/8 + lp_b0
        v8f acc[4];
#pragma unroll
        for (int t = 0; t < 4; t++) acc[t] = vzero8();
#pragma unroll
        for (int kc = 0; kc < 2; kc++) {
            v16h a = load_a_frag(ScA, 64, 32*kc, lane);
#pragma unroll
            for (int nt = 0; nt < 4; nt++)
                acc[nt] = wmma16(a, load_b_frag(LP0pk, kc*4 + nt, lane), acc[nt]);
        }
#pragma unroll
        for (int nt = 0; nt < 4; nt++) {
            int col = 16*nt + n;
            float bias = lp_b0v[col];
#pragma unroll
            for (int v = 0; v < 8; v++) {
                int row = blockIdx.x*16 + v + 8*hh;
                if (row < nA) out[(size_t)row*240 + col] = acc[nt][v] + bias;
            }
        }
    }
    {   // y1: per vector component, g1[:,:,mc][16x32] @ lp_w1
#pragma unroll
        for (int mc = 0; mc < 3; mc++) {
            v16h a = load_a_frag(G1A + mc*512, 32, 0, lane);
#pragma unroll
            for (int nt = 0; nt < 2; nt++) {
                v8f acc = wmma16(a, load_b_frag(LP1pk, nt, lane), vzero8());
                int col = 64 + (16*nt + n)*3 + mc;
#pragma unroll
                for (int v = 0; v < 8; v++) {
                    int row = blockIdx.x*16 + v + 8*hh;
                    if (row < nA) out[(size_t)row*240 + col] = acc[v];
                }
            }
        }
    }
    {   // y2: per component, g2[:,:,mc][16x16 pad 32] @ lp_w2
#pragma unroll
        for (int mc = 0; mc < 5; mc++) {
            v16h a = load_a_frag(G2A + mc*512, 32, 0, lane);
            v8f acc = wmma16(a, load_b_frag(LP2pk, 0, lane), vzero8());
            int col = 160 + n*5 + mc;
#pragma unroll
            for (int v = 0; v < 8; v++) {
                int row = blockIdx.x*16 + v + 8*hh;
                if (row < nA) out[(size_t)row*240 + col] = acc[v];
            }
        }
    }
}

//==================== host launcher =====================
extern "C" void kernel_launch(void* const* d_in, const int* in_sizes, int n_in,
                              void* d_out, int out_size, void* d_ws, size_t ws_size,
                              hipStream_t stream)
{
    (void)n_in; (void)out_size; (void)ws_size;
    const float* latents = (const float*)d_in[0];
    const float* nodef   = (const float*)d_in[1];
    const float* edgef   = (const float*)d_in[2];
    const float* wD      = (const float*)d_in[4];
    const float* ln_n_w0 = (const float*)d_in[5];
    const float* ln_n_b0 = (const float*)d_in[6];
    const float* ln_n_w1 = (const float*)d_in[7];
    const float* ln_n_w2 = (const float*)d_in[8];
    const float* ln_e_w0 = (const float*)d_in[9];
    const float* ln_e_b0 = (const float*)d_in[10];
    const float* ln_e_w1 = (const float*)d_in[11];
    const float* ln_e_w2 = (const float*)d_in[12];
    const float* so2_w0  = (const float*)d_in[13];
    const float* so2_w1r = (const float*)d_in[14];
    const float* so2_w1i = (const float*)d_in[15];
    const float* so2_w2r = (const float*)d_in[16];
    const float* so2_w2i = (const float*)d_in[17];
    const float* lp_w0   = (const float*)d_in[18];
    const float* lp_b0v  = (const float*)d_in[19];
    const float* lp_w1   = (const float*)d_in[20];
    const float* lp_w2   = (const float*)d_in[21];
    const int*   edge_index = (const int*)d_in[22];
    const int*   active  = (const int*)d_in[23];

    int E_total = in_sizes[22] / 2;   // edge_index (2, E)
    int nA      = in_sizes[23];       // active_edges

    unsigned char* ws = (unsigned char*)d_ws;
    h16* W0pk  = (h16*)(ws + 0);        // 480x160 -> 153600 B
    h16* W1Ppk = (h16*)(ws + 153600);   // 288x48  -> 27648 B
    h16* W1Npk = (h16*)(ws + 181248);   // 288x48  -> 27648 B
    h16* W2Ppk = (h16*)(ws + 208896);   // 96x16   -> 3072 B
    h16* W2Npk = (h16*)(ws + 211968);   // 96x16   -> 3072 B
    h16* LP0pk = (h16*)(ws + 215040);   // 64x64   -> 8192 B
    h16* LP1pk = (h16*)(ws + 223232);   // 32x32   -> 2048 B
    h16* LP2pk = (h16*)(ws + 225280);   // 32x16   -> 1024 B

    auto pack = [&](const float* A, int KA, const float* B, int KB,
                    float sA, float sB, int N, int Kp, h16* dst) {
        int total  = (Kp / 32) * (N / 16) * 512;
        int blocks = (total + 255) / 256;
        pack_weights_kernel<<<blocks, 256, 0, stream>>>(A, KA, B, KB, sA, sB, N, Kp, dst);
    };

    const float i0 = 1.0f / sqrtf(464.0f);
    const float i1 = 1.0f / 12.0f;            // 1/sqrt(144)
    const float i2 = 1.0f / sqrtf(48.0f);
    pack(so2_w0, 464, nullptr, 0, i0, 0.0f, 160, 480, W0pk);
    pack(so2_w1r, 144, so2_w1i, 144, i1, -i1, 48, 288, W1Ppk);  // op1 = p1*w1r - n1*w1i
    pack(so2_w1i, 144, so2_w1r, 144, i1,  i1, 48, 288, W1Npk);  // on1 = p1*w1i + n1*w1r
    pack(so2_w2r, 48, so2_w2i, 48, i2, -i2, 16, 96, W2Ppk);
    pack(so2_w2i, 48, so2_w2r, 48, i2,  i2, 16, 96, W2Npk);
    pack(lp_w0, 64, nullptr, 0, 0.125f, 0.0f, 64, 64, LP0pk);   // 1/sqrt(64)
    pack(lp_w1, 32, nullptr, 0, 1.0f / sqrtf(32.0f), 0.0f, 32, 32, LP1pk);
    pack(lp_w2, 16, nullptr, 0, 0.25f, 0.0f, 16, 32, LP2pk);    // 1/sqrt(16), K padded to 32

    int tiles = (nA + 15) / 16;
    edge_layer_kernel<<<tiles, 32, 0, stream>>>(
        latents, nodef, edgef, wD,
        ln_n_w0, ln_n_b0, ln_n_w1, ln_n_w2,
        ln_e_w0, ln_e_b0, ln_e_w1, ln_e_w2,
        lp_b0v, edge_index, active, E_total, nA,
        W0pk, W1Ppk, W1Npk, W2Ppk, W2Npk, LP0pk, LP1pk, LP2pk,
        (float*)d_out);
}
